// RNN_34024730919141
// MI455X (gfx1250) — matrix operand order repaired and checked by reading, not compiled, not run on hardware
//
#include <hip/hip_runtime.h>
#include <hip/hip_bf16.h>

// ---------------------------------------------------------------------------
// GRU (B=4096, T=140, E=128, H=256, O=3) for gfx1250 (CDNA5, wave32, WMMA)
//
// Strategy:
//  K0: P[v][g] = embed[v] . W_ih[g] + b_ih[g] + b_hh[g]   (401 x 768 fp32)
//      -> embedding+input-projection collapses to a table gather.
//  K1: pack W_hh (fp32 [768][256]) into bf16 WMMA B-fragments:
//      fragment(nt,kt) element (lane, j) = W_hh[nt*16+lane%16][kt*32+8*(lane/16)+j+(j>=8?8:0)]
//  K2: persistent GRU kernel: 256 WGs x 768 threads (24 waves), 16 batch rows
//      per WG, full t-loop in-kernel. Each wave holds its W_hh B-fragments
//      (2 n-tiles x 8 k-tiles = 128 VGPRs) in registers for all 140 steps.
// ---------------------------------------------------------------------------

typedef __attribute__((ext_vector_type(16))) __bf16       v16bf;
typedef __attribute__((ext_vector_type(8)))  float        v8f;
typedef __attribute__((ext_vector_type(4)))  unsigned int v4u;

union Frag16 {
    v16bf v;
    v4u   q[2];
};

__device__ __forceinline__ unsigned short f2bf(float f) {
    unsigned int u = __float_as_uint(f);
    unsigned int r = u + 0x7FFFu + ((u >> 16) & 1u);   // round-to-nearest-even
    return (unsigned short)(r >> 16);
}

// ---------------- K0: P = embed @ W_ih^T + b_ih + b_hh  (401 x 768) --------
__global__ void k_precompute_P(const float* __restrict__ embed,
                               const float* __restrict__ W_ih,
                               const float* __restrict__ b_ih,
                               const float* __restrict__ b_hh,
                               float* __restrict__ P) {
    int idx = blockIdx.x * blockDim.x + threadIdx.x;
    if (idx >= 401 * 768) return;
    int v = idx / 768, g = idx % 768;
    const float* er = embed + v * 128;
    const float* wr = W_ih + g * 128;
    float s = b_ih[g] + b_hh[g];
#pragma unroll 8
    for (int e = 0; e < 128; ++e) s += er[e] * wr[e];
    P[idx] = s;
}

// ---------------- K1: pack W_hh into bf16 WMMA B-fragments -----------------
// Bpack[((nt*8+kt)*32 + lane)*16 + j] = bf16(W_hh[nt*16 + (lane&15)][kt*32 + 8*(lane>>4) + j + (j>=8?8:0)])
__global__ void k_pack_whh(const float* __restrict__ W_hh,
                           unsigned short* __restrict__ Bpack) {
    int idx = blockIdx.x * blockDim.x + threadIdx.x;
    if (idx >= 48 * 8 * 32 * 16) return;
    int j    = idx & 15;
    int lane = (idx >> 4) & 31;
    int kt   = (idx >> 9) & 7;
    int nt   = idx >> 12;
    Bpack[idx] = f2bf(W_hh[(nt * 16 + (lane & 15)) * 256 + (kt * 32 + ((lane >> 4) << 3) + j + ((j >> 3) << 3))]);
}

// ---------------- K2: persistent GRU recurrence ----------------------------
__launch_bounds__(768, 1)
__global__ void k_gru(const int*   __restrict__ input,
                      const float* __restrict__ hidden,
                      const float* __restrict__ P,
                      const unsigned short* __restrict__ Bpack,
                      const float* __restrict__ W_out,
                      const float* __restrict__ b_out,
                      float* __restrict__ out_logp,
                      float* __restrict__ out_h) {
    __shared__ __align__(16) float          gh[16 * 768];    // 48 KB
    __shared__ __align__(16) float          h32[16 * 256];   // 16 KB
    __shared__ __align__(16) unsigned short hbf[16 * 256];   //  8 KB
    __shared__               int            toks[16 * 140];  //  9 KB

    const int tid  = threadIdx.x;
    const int lane = tid & 31;
    const int wave = tid >> 5;            // 0..23
    const int b0   = blockIdx.x * 16;     // batch row base

    // init hidden state (fp32 master + bf16 shadow)
    for (int i = tid; i < 16 * 256; i += 768) {
        float h = hidden[(b0 + (i >> 8)) * 256 + (i & 255)];
        h32[i] = h;
        hbf[i] = f2bf(h);
    }
    // stage all tokens for this WG's rows
    for (int i = tid; i < 16 * 140; i += 768) {
        toks[i] = input[(b0 + i / 140) * 140 + (i % 140)];
    }

    // load this wave's W_hh B-fragments once: 2 n-tiles x 8 k-tiles
    v16bf Bf[16];
    {
        const v16bf* bp = reinterpret_cast<const v16bf*>(Bpack);
        const int nt0 = wave * 2;
#pragma unroll
        for (int kt = 0; kt < 8; ++kt) {
            Bf[kt]     = bp[(nt0 * 8 + kt) * 32 + lane];
            Bf[8 + kt] = bp[((nt0 + 1) * 8 + kt) * 32 + lane];
        }
    }
    __syncthreads();

    // A-fragment (16x32 bf16) addressing: lane -> M = lane&15,
    // element j: K = kt*32 + (lane>=16 ? 8 : 0) + (j<8 ? j : 8 + j)  (two
    // contiguous 8-halfword runs -> two 16B LDS loads).
    const int mrow  = lane & 15;
    const int base8 = (lane >> 4) << 3;           // 0 or 8
    // C/D fragment: lane -> N = lane&15, vgpr v -> M = v + (lane>=16 ? 8 : 0)
    const int crow0 = (lane >> 4) << 3;
    const int ccol  = lane & 15;
    const int n0    = wave * 32;                  // this wave's gate-col base

    for (int t = 0; t < 140; ++t) {
        // ---- gh(tile) = h @ W_hh^T via WMMA bf16 -> fp32 ----
        v8f acc0 = {};
        v8f acc1 = {};
#pragma unroll
        for (int kt = 0; kt < 8; ++kt) {
            Frag16 a;
            a.q[0] = *(const v4u*)&hbf[mrow * 256 + kt * 32 + base8];
            a.q[1] = *(const v4u*)&hbf[mrow * 256 + kt * 32 + base8 + 16];
            acc0 = __builtin_amdgcn_wmma_f32_16x16x32_bf16(
                false, a.v, false, Bf[kt],     (short)0, acc0, false, false);
            acc1 = __builtin_amdgcn_wmma_f32_16x16x32_bf16(
                false, a.v, false, Bf[8 + kt], (short)0, acc1, false, false);
        }
#pragma unroll
        for (int v = 0; v < 8; ++v) {
            gh[(crow0 + v) * 768 + n0 + ccol]      = acc0[v];
            gh[(crow0 + v) * 768 + n0 + 16 + ccol] = acc1[v];
        }
        __syncthreads();

        // ---- elementwise GRU gates + state update ----
        for (int i = tid; i < 16 * 256; i += 768) {
            int m = i >> 8, j = i & 255;
            const float* pr = P + toks[m * 140 + t] * 768;
            float i_r = pr[j], i_z = pr[256 + j], i_n = pr[512 + j];
            float h_r = gh[m * 768 + j];
            float h_z = gh[m * 768 + 256 + j];
            float h_n = gh[m * 768 + 512 + j];
            float r  = 1.0f / (1.0f + __expf(-(i_r + h_r)));
            float z  = 1.0f / (1.0f + __expf(-(i_z + h_z)));
            float n  = tanhf(i_n + r * h_n);
            float h  = h32[i];
            float hn = (1.0f - z) * n + z * h;
            h32[i] = hn;
            hbf[i] = f2bf(hn);
        }
        __syncthreads();
    }

    // ---- classifier head + log_softmax (O=3, tiny) ----
    if (tid < 16) {
        const int m = tid;
        float l[3];
#pragma unroll
        for (int o = 0; o < 3; ++o) {
            float s = b_out[o];
            for (int j = 0; j < 256; ++j) s += W_out[o * 256 + j] * h32[m * 256 + j];
            l[o] = s;
        }
        float mx  = fmaxf(l[0], fmaxf(l[1], l[2]));
        float se  = __expf(l[0] - mx) + __expf(l[1] - mx) + __expf(l[2] - mx);
        float lse = mx + __logf(se);
#pragma unroll
        for (int o = 0; o < 3; ++o) out_logp[(b0 + m) * 3 + o] = l[o] - lse;
    }
    for (int i = tid; i < 16 * 256; i += 768) {
        out_h[(b0 + (i >> 8)) * 256 + (i & 255)] = h32[i];
    }
}

// ---------------------------------------------------------------------------
extern "C" void kernel_launch(void* const* d_in, const int* in_sizes, int n_in,
                              void* d_out, int out_size, void* d_ws, size_t ws_size,
                              hipStream_t stream) {
    const int*   input  = (const int*)d_in[0];
    const float* hidden = (const float*)d_in[1];
    const float* embed  = (const float*)d_in[2];
    const float* W_ih   = (const float*)d_in[3];
    const float* W_hh   = (const float*)d_in[4];
    const float* b_ih   = (const float*)d_in[5];
    const float* b_hh   = (const float*)d_in[6];
    const float* W_out  = (const float*)d_in[7];
    const float* b_out  = (const float*)d_in[8];

    float* out = (float*)d_out;                 // [B*3 logp] ++ [B*256 h_final]

    // workspace: P (401*768 fp32 = 1,231,872 B, 256-aligned size) then Bpack
    float*          P     = (float*)d_ws;
    unsigned short* Bpack = (unsigned short*)((char*)d_ws + 401 * 768 * 4);

    k_precompute_P<<<(401 * 768 + 255) / 256, 256, 0, stream>>>(
        embed, W_ih, b_ih, b_hh, P);
    k_pack_whh<<<(48 * 8 * 32 * 16 + 255) / 256, 256, 0, stream>>>(W_hh, Bpack);
    k_gru<<<4096 / 16, 768, 0, stream>>>(
        input, hidden, P, Bpack, W_out, b_out, out, out + 4096 * 3);
}
